// MHSA_22677427323424
// MI455X (gfx1250) — matrix operand order repaired and checked by reading, not compiled, not run on hardware
//
#include <hip/hip_runtime.h>

typedef _Float16 f16;
typedef __attribute__((ext_vector_type(16))) _Float16 v16h;
typedef __attribute__((ext_vector_type(8)))  _Float16 v8h;
typedef __attribute__((ext_vector_type(8)))  float    v8f;

union AFrag { v16h v; v8h h[2]; };

#define WMMA_F16(a, b, c) \
  __builtin_amdgcn_wmma_f32_16x16x32_f16(false, (a), false, (b), (short)0, (c), false, false)

static constexpr int Bx = 4, Nx = 2048, Dx = 768, Hx = 12, HD = 64;
static constexpr int Mrows = Bx * Nx;          // 8192
static constexpr int QKVN  = 3 * Dx;           // 2304

// A-fragment load: row per lane (lane%16), contiguous k = 16*half + s (matches B)
__device__ __forceinline__ void load_a(AFrag& a, const f16* row, int k0, int half) {
  a.h[0] = *(const v8h*)(row + k0 + half * 16);
  a.h[1] = *(const v8h*)(row + k0 + half * 16 + 8);
}

// ---------------- conversion / transpose kernels ----------------

__global__ __launch_bounds__(256) void cvt_f32_to_f16(const float* __restrict__ in,
                                                      f16* __restrict__ out, int n) {
  int i = blockIdx.x * 256 + threadIdx.x;
  if (i < n) out[i] = (f16)in[i];
}

// out[c*R + r] = (f16) in[r*C + c]   (B stored [n][k]: WMMA B-fragments contiguous)
__global__ __launch_bounds__(256) void cvt_transpose(const float* __restrict__ in,
                                                     f16* __restrict__ out, int R, int C) {
  int idx = blockIdx.x * 256 + threadIdx.x;
  if (idx >= R * C) return;
  int r = idx % R;
  int c = idx / R;
  out[idx] = (f16)in[(size_t)r * C + c];
}

// ---------------- shared 32x64 wave-tile GEMM core (ping-pong pipelined) ----------------

__device__ __forceinline__ void gemm_core(const f16* arow0, const f16* arow1,
                                          const f16* bcol, int half, v8f acc[2][4]) {
  AFrag a0[2], a1[2];
  v16h b[2][4];
  load_a(a0[0], arow0, 0, half);
  load_a(a1[0], arow1, 0, half);
#pragma unroll
  for (int t = 0; t < 4; ++t) b[0][t] = *(const v16h*)(bcol + (size_t)(t * 16) * Dx);

  for (int k0 = 0; k0 < Dx; k0 += 64) {
    // prefetch buffer 1 at k0+32 (always in range: 64 | 768)
    load_a(a0[1], arow0, k0 + 32, half);
    load_a(a1[1], arow1, k0 + 32, half);
#pragma unroll
    for (int t = 0; t < 4; ++t)
      b[1][t] = *(const v16h*)(bcol + (size_t)(t * 16) * Dx + k0 + 32);

#pragma unroll
    for (int t = 0; t < 4; ++t) {
      acc[0][t] = WMMA_F16(a0[0].v, b[0][t], acc[0][t]);
      acc[1][t] = WMMA_F16(a1[0].v, b[0][t], acc[1][t]);
    }

    if (k0 + 64 < Dx) {                       // prefetch buffer 0 at k0+64
      load_a(a0[0], arow0, k0 + 64, half);
      load_a(a1[0], arow1, k0 + 64, half);
#pragma unroll
      for (int t = 0; t < 4; ++t)
        b[0][t] = *(const v16h*)(bcol + (size_t)(t * 16) * Dx + k0 + 64);
    }

#pragma unroll
    for (int t = 0; t < 4; ++t) {
      acc[0][t] = WMMA_F16(a0[1].v, b[1][t], acc[0][t]);
      acc[1][t] = WMMA_F16(a1[1].v, b[1][t], acc[1][t]);
    }
  }
}

// ---------------- QKV GEMM: [8192 x 768] x [768 x 2304] -> Q/K/Vt (f16) ----------------

__global__ __launch_bounds__(256)
void qkv_gemm(const f16* __restrict__ xh, const f16* __restrict__ wT,
              f16* __restrict__ Qh, f16* __restrict__ Kh, f16* __restrict__ Vt) {
  const int wave = threadIdx.x >> 5;
  const int lane = threadIdx.x & 31;
  const int half = lane >> 4;
  const int lm   = lane & 15;

  const int lin = blockIdx.x * 8 + wave;      // 9216 wave-tiles
  const int j   = lin % (QKVN / 64);          // 64-wide column tile (0..35)
  const int mt  = lin / (QKVN / 64);          // 32-row tile (0..255)
  const int m0  = mt * 32;

  v8f acc[2][4] = {};
  gemm_core(xh + (size_t)(m0 + lm) * Dx, xh + (size_t)(m0 + 16 + lm) * Dx,
            wT + (size_t)(j * 64 + lm) * Dx + half * 16, half, acc);

  const int mat = j / Hx;        // 0=Q, 1=K, 2=V
  const int h   = j % Hx;
  const int b_  = m0 >> 11;
  const int q0  = m0 & 2047;
  const size_t bh = (size_t)b_ * Hx + h;

  if (mat == 0) {
    f16* dst = Qh + bh * Nx * HD;              // fold 1/sqrt(64) into Q
#pragma unroll
    for (int r = 0; r < 2; ++r)
#pragma unroll
      for (int t = 0; t < 4; ++t)
#pragma unroll
        for (int g = 0; g < 8; ++g)
          dst[(size_t)(q0 + r * 16 + g + 8 * half) * HD + t * 16 + lm] =
              (f16)(acc[r][t][g] * 0.125f);
  } else if (mat == 1) {
    f16* dst = Kh + bh * Nx * HD;
#pragma unroll
    for (int r = 0; r < 2; ++r)
#pragma unroll
      for (int t = 0; t < 4; ++t)
#pragma unroll
        for (int g = 0; g < 8; ++g)
          dst[(size_t)(q0 + r * 16 + g + 8 * half) * HD + t * 16 + lm] = (f16)acc[r][t][g];
  } else {
    f16* dst = Vt + bh * HD * Nx;              // [d][n]
#pragma unroll
    for (int r = 0; r < 2; ++r)
#pragma unroll
      for (int t = 0; t < 4; ++t)
#pragma unroll
        for (int g = 0; g < 8; ++g)
          dst[(size_t)(t * 16 + lm) * Nx + q0 + r * 16 + g + 8 * half] = (f16)acc[r][t][g];
  }
}

// ---------------- flash attention: one wave per 16-row Q tile ----------------

__device__ __forceinline__ void load_k_frags(v16h* kb, const f16* kbase, int n0,
                                             int lm, int half) {
  const f16* kr0 = kbase + (size_t)(n0 + lm) * HD + half * 16;
  kb[0] = *(const v16h*)(kr0);
  kb[1] = *(const v16h*)(kr0 + 32);
  const f16* kr1 = kbase + (size_t)(n0 + 16 + lm) * HD + half * 16;
  kb[2] = *(const v16h*)(kr1);
  kb[3] = *(const v16h*)(kr1 + 32);
}

__global__ __launch_bounds__(256)
void attn_fwd(const f16* __restrict__ Qh, const f16* __restrict__ Kh,
              const f16* __restrict__ Vt, f16* __restrict__ Oh) {
  __shared__ f16 pbuf[8][16 * 32];            // per-wave P transpose tile

  const int wave = threadIdx.x >> 5;
  const int lane = threadIdx.x & 31;
  const int half = lane >> 4;
  const int lm   = lane & 15;

  const int lin = blockIdx.x * 8 + wave;      // 6144 wave-tiles
  const int qt  = lin & 127;
  const int bh  = lin >> 7;                   // batch*head (0..47)

  const f16* qbase = Qh + ((size_t)bh * Nx + qt * 16 + lm) * HD;
  AFrag qa0, qa1;                             // Q 16x64 as two K=32 A-fragments
  load_a(qa0, qbase, 0, half);
  load_a(qa1, qbase, 32, half);

  const f16* kbase = Kh + (size_t)bh * Nx * HD;
  const f16* vbase = Vt + (size_t)bh * HD * Nx;
  f16* pl = pbuf[wave];

  v8f o[4] = {};
  float m[8], l[8];
#pragma unroll
  for (int g = 0; g < 8; ++g) { m[g] = -1.0e30f; l[g] = 0.0f; }

  // one 32-key stage: S-WMMA -> V loads -> online softmax -> P transpose -> PV-WMMA
  auto stage = [&](const v16h* kbf, int ns) {
    v8f s0 = {}, s1 = {};
    s0 = WMMA_F16(qa0.v, kbf[0], s0);
    s0 = WMMA_F16(qa1.v, kbf[1], s0);
    s1 = WMMA_F16(qa0.v, kbf[2], s1);
    s1 = WMMA_F16(qa1.v, kbf[3], s1);

    v16h vb[4];                               // in flight during softmax
#pragma unroll
    for (int t = 0; t < 4; ++t)
      vb[t] = *(const v16h*)(vbase + (size_t)(t * 16 + lm) * Nx + ns + half * 16);

    float sc[8];
#pragma unroll
    for (int g = 0; g < 8; ++g) {
      float a = s0[g], bb = s1[g];
      float mx = fmaxf(a, bb);
      mx = fmaxf(mx, __shfl_xor(mx, 1, 32));
      mx = fmaxf(mx, __shfl_xor(mx, 2, 32));
      mx = fmaxf(mx, __shfl_xor(mx, 4, 32));
      mx = fmaxf(mx, __shfl_xor(mx, 8, 32));
      float mn = fmaxf(m[g], mx);
      float scale = __expf(m[g] - mn);
      float e0 = __expf(a - mn);
      float e1 = __expf(bb - mn);
      float sum = e0 + e1;
      sum += __shfl_xor(sum, 1, 32);
      sum += __shfl_xor(sum, 2, 32);
      sum += __shfl_xor(sum, 4, 32);
      sum += __shfl_xor(sum, 8, 32);
      l[g] = l[g] * scale + sum;
      m[g] = mn;
      sc[g] = scale;
      const int row = g + 8 * half;
      pl[row * 32 + lm]      = (f16)e0;
      pl[row * 32 + 16 + lm] = (f16)e1;
    }
#pragma unroll
    for (int t = 0; t < 4; ++t)
#pragma unroll
      for (int g = 0; g < 8; ++g) o[t][g] *= sc[g];

    // same-wave LDS store -> load (C-layout -> A-layout transpose of P)
    asm volatile("s_wait_dscnt 0" ::: "memory");

    AFrag pa;
    pa.h[0] = *(const v8h*)(pl + lm * 32 + half * 16);
    pa.h[1] = *(const v8h*)(pl + lm * 32 + half * 16 + 8);

#pragma unroll
    for (int t = 0; t < 4; ++t) o[t] = WMMA_F16(pa.v, vb[t], o[t]);
  };

  v16h kb0[4], kb1[4];
  load_k_frags(kb0, kbase, 0, lm, half);
  for (int n0 = 0; n0 < Nx; n0 += 64) {       // ping-pong on K buffers, no reg copies
    load_k_frags(kb1, kbase, n0 + 32, lm, half);
    stage(kb0, n0);
    if (n0 + 64 < Nx) load_k_frags(kb0, kbase, n0 + 64, lm, half);
    stage(kb1, n0 + 32);
  }

  // ---- normalize and write O as [b][q][h*64+d] f16 for the proj GEMM ----
  const int b = bh / Hx, h = bh % Hx;
#pragma unroll
  for (int g = 0; g < 8; ++g) {
    const float rl = 1.0f / l[g];
    const int q = qt * 16 + g + 8 * half;
    f16* dst = Oh + ((size_t)(b * Nx + q)) * Dx + h * HD + lm;
#pragma unroll
    for (int t = 0; t < 4; ++t) dst[t * 16] = (f16)(o[t][g] * rl);
  }
}

// ---------------- proj GEMM: [8192 x 768] x [768 x 768] + bias -> f32 out ----------------

__global__ __launch_bounds__(256)
void proj_gemm(const f16* __restrict__ Oh, const f16* __restrict__ wT,
               const float* __restrict__ bias, float* __restrict__ out) {
  const int wave = threadIdx.x >> 5;
  const int lane = threadIdx.x & 31;
  const int half = lane >> 4;
  const int lm   = lane & 15;

  const int lin = blockIdx.x * 8 + wave;      // 3072 wave-tiles
  const int j   = lin % (Dx / 64);            // 0..11
  const int mt  = lin / (Dx / 64);            // 0..255
  const int m0  = mt * 32;

  v8f acc[2][4] = {};
  gemm_core(Oh + (size_t)(m0 + lm) * Dx, Oh + (size_t)(m0 + 16 + lm) * Dx,
            wT + (size_t)(j * 64 + lm) * Dx + half * 16, half, acc);

#pragma unroll
  for (int t = 0; t < 4; ++t) {
    const int n = j * 64 + t * 16 + lm;
    const float bn = bias[n];
#pragma unroll
    for (int r = 0; r < 2; ++r)
#pragma unroll
      for (int g = 0; g < 8; ++g)
        out[(size_t)(m0 + r * 16 + g + 8 * half) * Dx + n] = acc[r][t][g] + bn;
  }
}

// ---------------- launch ----------------

extern "C" void kernel_launch(void* const* d_in, const int* in_sizes, int n_in,
                              void* d_out, int out_size, void* d_ws, size_t ws_size,
                              hipStream_t stream) {
  (void)in_sizes; (void)n_in; (void)out_size; (void)ws_size;
  const float* x      = (const float*)d_in[0];
  const float* w_qkv  = (const float*)d_in[1];
  const float* w_proj = (const float*)d_in[2];
  const float* b_proj = (const float*)d_in[3];
  float* out = (float*)d_out;

  char* ws = (char*)d_ws;
  f16* xh     = (f16*)(ws);                                  // 8192*768
  f16* wqkvT  = (f16*)(ws + 12582912);                       // 2304*768
  f16* wprojT = (f16*)(ws + 12582912 + 3538944);             // 768*768
  f16* Qh     = (f16*)(ws + 17301504);                       // 48*2048*64
  f16* Kh     = (f16*)(ws + 17301504 + 12582912);
  f16* Vt     = (f16*)(ws + 17301504 + 2 * 12582912);
  f16* Oh     = (f16*)(ws + 17301504 + 3 * 12582912);        // total ~64.5 MB

  cvt_f32_to_f16<<<(Mrows * Dx) / 256, 256, 0, stream>>>(x, xh, Mrows * Dx);
  cvt_transpose<<<(Dx * QKVN) / 256, 256, 0, stream>>>(w_qkv, wqkvT, Dx, QKVN);
  cvt_transpose<<<(Dx * Dx) / 256, 256, 0, stream>>>(w_proj, wprojT, Dx, Dx);

  qkv_gemm<<<(Mrows / 32) * (QKVN / 64) / 8, 256, 0, stream>>>(xh, wqkvT, Qh, Kh, Vt);
  attn_fwd<<<(Bx * Hx) * (Nx / 16) / 8, 256, 0, stream>>>(Qh, Kh, Vt, Oh);
  proj_gemm<<<(Mrows / 32) * (Dx / 64) / 8, 256, 0, stream>>>(Oh, wprojT, b_proj, out);
}
